// VectorQuantizer_27934467293548
// MI455X (gfx1250) — matrix operand order repaired and checked by reading, not compiled, not run on hardware
//
#include <hip/hip_runtime.h>
#include <hip/hip_bf16.h>
#include <math.h>
#include <stdint.h>

#define NUM_EMB 2048
#define EMB_DIM 512
#define BATCH   64
#define TLEN    256
#define NROWS   (BATCH * TLEN)           // 16384
#define BDT     (BATCH * EMB_DIM * TLEN) // 8388608
#define NTILES  (NUM_EMB / 16)           // 128 code tiles

// ---- CDNA5 WMMA types -------------------------------------------------------
typedef __attribute__((ext_vector_type(16))) __bf16      bf16x16;
typedef __attribute__((ext_vector_type(8)))  float       floatx8;
typedef __attribute__((ext_vector_type(4)))  unsigned int uintx4;

union FragB16 { uintx4 u[2]; bf16x16 v; };   // 32 bytes = 16 bf16 = one WMMA operand

// Split fp32 into bf16 hi + bf16 lo (round-to-nearest-even both times).
__device__ __forceinline__ void split_bf16(float f, unsigned short& hi, unsigned short& lo) {
  union { float f; unsigned u; } a; a.f = f;
  unsigned r = a.u + 0x7FFFu + ((a.u >> 16) & 1u);
  hi = (unsigned short)(r >> 16);
  union { unsigned u; float f; } h; h.u = ((unsigned)hi) << 16;
  float rem = f - h.f;
  union { float f; unsigned u; } b2; b2.f = rem;
  unsigned r2 = b2.u + 0x7FFFu + ((b2.u >> 16) & 1u);
  lo = (unsigned short)(r2 >> 16);
}

// Branchless sorted top-3 (smallest) insertion.
__device__ __forceinline__ void top3_insert(float d, unsigned c,
                                            float& b0, float& b1, float& b2,
                                            unsigned& i0, unsigned& i1, unsigned& i2) {
  bool lt0 = d < b0, lt1 = d < b1, lt2 = d < b2;
  float    nb2 = lt1 ? b1 : (lt2 ? d : b2);
  unsigned ni2 = lt1 ? i1 : (lt2 ? c : i2);
  float    nb1 = lt0 ? b0 : (lt1 ? d : b1);
  unsigned ni1 = lt0 ? i0 : (lt1 ? c : i1);
  float    nb0 = lt0 ? d  : b0;
  unsigned ni0 = lt0 ? c  : i0;
  b0 = nb0; b1 = nb1; b2 = nb2;
  i0 = ni0; i1 = ni1; i2 = ni2;
}

// ---- Kernel 1: codebook prep (bf16 hi/lo split + squared norms) ------------
__global__ void __launch_bounds__(256)
vq_prep_codes(const float* __restrict__ W,
              unsigned short* __restrict__ whi,
              unsigned short* __restrict__ wlo,
              float* __restrict__ cnorm) {
  int row  = blockIdx.x * 8 + (threadIdx.x >> 5);
  int lane = threadIdx.x & 31;
  const float* wr = W + row * EMB_DIM;
  float s = 0.f;
  for (int j = 0; j < EMB_DIM / 32; ++j) {
    int d = lane + 32 * j;
    float v = wr[d];
    s += v * v;
    unsigned short h, l;
    split_bf16(v, h, l);
    whi[row * EMB_DIM + d] = h;
    wlo[row * EMB_DIM + d] = l;
  }
  for (int off = 16; off >= 1; off >>= 1) s += __shfl_xor(s, off, 32);
  if (lane == 0) cnorm[row] = s;
}

// ---- Kernel 2: zero accumulators (graph-capture-safe init) -----------------
__global__ void vq_init(float* __restrict__ counts, float* __restrict__ sse) {
  int tid = blockIdx.x * blockDim.x + threadIdx.x;
  if (tid < NUM_EMB) counts[tid] = 0.f;
  if (tid == 0) *sse = 0.f;
}

// ---- Kernel 3: WMMA GEMM + fused top-3 -------------------------------------
// 32 rows per block, 2 waves (16 rows each). A staged once in LDS as split
// bf16. B (16 codes x 512 dims, hi+lo) double-buffered in LDS, filled by
// GLOBAL_LOAD_ASYNC_TO_LDS_B128 (ASYNCcnt) so L2 latency overlaps the WMMAs.
#define GB_ROWS    32
#define GB_THREADS 64
#define PITCH_H    520      // bf16 per padded LDS row (1040 B, 16B-aligned)
#define PITCH_U    260
#define SM_AHI     0        // 32*520*2 = 33280
#define SM_ALO     33280    // 33280
#define SM_B0      66560    // two buffers of (hi 16640 + lo 16640) = 2*33280
#define SM_MVAL    133120   // 32*16*3*4 = 6144
#define SM_MIDX    139264   // 6144
#define SMEM_BYTES 145408

__global__ void __launch_bounds__(GB_THREADS)
vq_gemm_top3(const float* __restrict__ x,
             const unsigned short* __restrict__ whi,
             const unsigned short* __restrict__ wlo,
             const float* __restrict__ cnorm,
             unsigned* __restrict__ idx3) {
  extern __shared__ char smem[];
  unsigned short* sAhi = (unsigned short*)(smem + SM_AHI);
  unsigned short* sAlo = (unsigned short*)(smem + SM_ALO);
  float*          mval = (float*)(smem + SM_MVAL);
  unsigned*       midx = (unsigned*)(smem + SM_MIDX);

  const int tid    = threadIdx.x;
  const int lane   = tid & 31;
  const int waveId = tid >> 5;
  const int half   = lane >> 4;     // 16-lane half of the wave
  const int cLocal = lane & 15;     // code column within tile / A row within wave
  const int nBase  = blockIdx.x * GB_ROWS;
  const int b      = nBase >> 8;    // 32 consecutive tokens stay within one batch
  const int t0     = nBase & 255;

  // Stage A: gather x[b, d, t] (transposed view) -> split bf16 in LDS.
  for (int i = tid; i < GB_ROWS * EMB_DIM; i += GB_THREADS) {
    int r = i & (GB_ROWS - 1);
    int d = i >> 5;
    float v = x[b * (EMB_DIM * TLEN) + d * TLEN + t0 + r];
    unsigned short h, l;
    split_bf16(v, h, l);
    sAhi[r * PITCH_H + d] = h;
    sAlo[r * PITCH_H + d] = l;
  }

  // Per-lane top-3 state: slot v tracks row (v + 8*half), codes == cLocal mod 16.
  float    b0[8], b1[8], b2[8];
  unsigned i0[8], i1[8], i2[8];
#pragma unroll
  for (int v = 0; v < 8; ++v) { b0[v] = b1[v] = b2[v] = 3.4e38f; i0[v] = i1[v] = i2[v] = 0u; }

  const int aRow = waveId * 16 + cLocal;   // this lane's A matrix row (M = lane&15)
  const int aU   = aRow * PITCH_U;
  const int bbU  = cLocal * PITCH_U;

  // Async B-tile fill: wave0 streams hi, wave1 streams lo. 32 b128 per wave
  // per tile (16 codes x 1024 B each, padded +16 B per row in LDS).
  const unsigned smemBase = (unsigned)(uintptr_t)smem;  // low 32 bits = LDS offset
  const char* wSrc = (waveId == 0) ? (const char*)whi : (const char*)wlo;

  auto issue_tile = [&](int t) {
    unsigned ldsBuf = smemBase + SM_B0 + (unsigned)((t & 1) * 33280 + waveId * 16640);
    const char* gsrc = wSrc + (size_t)t * (16 * EMB_DIM * 2);
#pragma unroll 4
    for (int j = 0; j < 32; ++j) {
      int l = j * 32 + lane;                       // 0..1023: (code c = l>>6, du = l&63)
      unsigned laddr = ldsBuf + (unsigned)(l * 16 + (l >> 6) * 16);
      unsigned long long gaddr = (unsigned long long)(uintptr_t)(gsrc + l * 16);
      asm volatile("global_load_async_to_lds_b128 %0, %1, off"
                   :: "v"(laddr), "v"(gaddr)
                   : "memory");
    }
  };

  issue_tile(0);
  asm volatile("s_wait_asynccnt 0x0" ::: "memory");
  __syncthreads();   // A staged + B tile 0 resident

  for (int ct = 0; ct < NTILES; ++ct) {
    if (ct + 1 < NTILES) issue_tile(ct + 1);       // overlap with WMMAs below

    const char* bBase = smem + SM_B0 + (ct & 1) * 33280;  // hi at +0, lo at +16640
    const uintx4* qAhi = (const uintx4*)sAhi;
    const uintx4* qAlo = (const uintx4*)sAlo;
    const uintx4* qBhi = (const uintx4*)(bBase);
    const uintx4* qBlo = (const uintx4*)(bBase + 16640);

    floatx8 acc0, acc1, acc2;                      // 3 independent WMMA chains
#pragma unroll
    for (int v = 0; v < 8; ++v) { acc0[v] = 0.f; acc1[v] = 0.f; acc2[v] = 0.f; }

    for (int ks = 0; ks < EMB_DIM / 32; ++ks) {
      FragB16 aH, aL, bH, bL;
      // A 16x32 bf16 layout: lane<16 -> K {0..7,16..23}, lane>=16 -> K {8..15,24..31}
      int a0 = aU + ks * 16 + half * 4;            // uint index (16B-aligned)
      aH.u[0] = qAhi[a0 >> 2];
      aH.u[1] = qAhi[(a0 + 8) >> 2];
      aL.u[0] = qAlo[a0 >> 2];
      aL.u[1] = qAlo[(a0 + 8) >> 2];
      // B 32x16 bf16 layout: lane<16 -> K {0..7,16..23}, lane>=16 -> K {8..15,24..31}, N = lane&15
      int bb = bbU + ks * 16 + half * 4;
      bH.u[0] = qBhi[bb >> 2];
      bH.u[1] = qBhi[(bb + 8) >> 2];
      bL.u[0] = qBlo[bb >> 2];
      bL.u[1] = qBlo[(bb + 8) >> 2];
      // dot = hi*hi + hi*lo + lo*hi, each term on its own accumulator chain
      acc0 = __builtin_amdgcn_wmma_f32_16x16x32_bf16(false, aH.v, false, bH.v, (short)0, acc0, false, false);
      acc1 = __builtin_amdgcn_wmma_f32_16x16x32_bf16(false, aH.v, false, bL.v, (short)0, acc1, false, false);
      acc2 = __builtin_amdgcn_wmma_f32_16x16x32_bf16(false, aL.v, false, bH.v, (short)0, acc2, false, false);
    }

    unsigned code = ct * 16 + cLocal;
    float cn = cnorm[code];
#pragma unroll
    for (int v = 0; v < 8; ++v) {
      float key = cn - 2.0f * (acc0[v] + acc1[v] + acc2[v]);  // row-norm dropped
      top3_insert(key, code, b0[v], b1[v], b2[v], i0[v], i1[v], i2[v]);
    }

    asm volatile("s_wait_asynccnt 0x0" ::: "memory");  // next tile landed in LDS
    __syncthreads();                                   // everyone done reading cur
  }

  // Merge: each row has 16 disjoint candidate streams (one per lane column).
#pragma unroll
  for (int v = 0; v < 8; ++v) {
    int m    = v + 8 * half;                       // C layout: M = v + 8*(lane>>4)
    int row  = waveId * 16 + m;
    int base = (row * 16 + cLocal) * 3;
    mval[base + 0] = b0[v]; mval[base + 1] = b1[v]; mval[base + 2] = b2[v];
    midx[base + 0] = i0[v]; midx[base + 1] = i1[v]; midx[base + 2] = i2[v];
  }
  __syncthreads();

  if (tid < GB_ROWS) {
    float f0 = 3.4e38f, f1 = 3.4e38f, f2 = 3.4e38f;
    unsigned j0 = 0, j1 = 0, j2 = 0;
    int base = tid * 48;
    for (int j = 0; j < 48; ++j)
      top3_insert(mval[base + j], midx[base + j], f0, f1, f2, j0, j1, j2);
    unsigned n = nBase + tid;
    idx3[n * 3 + 0] = j0;
    idx3[n * 3 + 1] = j1;
    idx3[n * 3 + 2] = j2;
  }
}

// ---- Kernel 4: gather/average codes, write outputs, SSE + counts -----------
__global__ void __launch_bounds__(256)
vq_gather_outputs(const float* __restrict__ x, const float* __restrict__ W,
                  const unsigned* __restrict__ idx3,
                  float* __restrict__ out, float* __restrict__ counts,
                  float* __restrict__ sse) {
  __shared__ unsigned sI0[64], sI1[64], sI2[64];
  __shared__ float red[256];
  const int tid = threadIdx.x;
  const int b   = blockIdx.x >> 2;
  const int t0  = (blockIdx.x & 3) << 6;

  if (tid < 64) {
    unsigned n  = b * TLEN + t0 + tid;
    unsigned a0 = idx3[n * 3 + 0];
    unsigned a1 = idx3[n * 3 + 1];
    unsigned a2 = idx3[n * 3 + 2];
    sI0[tid] = a0; sI1[tid] = a1; sI2[tid] = a2;
    atomicAdd(&counts[a0], 1.0f);
    atomicAdd(&counts[a1], 1.0f);
    atomicAdd(&counts[a2], 1.0f);
    out[1 + BDT + 1 + n] = (float)a2;              // encoding_indices = idx[:, RANK-1]
  }
  __syncthreads();

  const int t = tid & 63;
  const unsigned q0 = sI0[t] * EMB_DIM, q1 = sI1[t] * EMB_DIM, q2 = sI2[t] * EMB_DIM;
  float accum = 0.f;
  for (int d = (tid >> 6); d < EMB_DIM; d += 4) {
    float xv = x[b * (EMB_DIM * TLEN) + d * TLEN + t0 + t];
    float q  = (W[q0 + d] + W[q1 + d] + W[q2 + d]) * (1.0f / 3.0f);
    float diff = q - xv;
    accum += diff * diff;
    // straight-through: inp + (quantized - inp), written back in [B, D, T]
    out[1 + b * (EMB_DIM * TLEN) + d * TLEN + t0 + t] = xv + (q - xv);
  }
  red[tid] = accum;
  __syncthreads();
  for (int s = 128; s > 0; s >>= 1) {
    if (tid < s) red[tid] += red[tid + s];
    __syncthreads();
  }
  if (tid == 0) atomicAdd(sse, red[0]);
}

// ---- Kernel 5: finalize loss + perplexity ----------------------------------
__global__ void __launch_bounds__(256)
vq_finalize(const float* __restrict__ counts, const float* __restrict__ sse,
            float* __restrict__ out) {
  __shared__ float red[256];
  const int tid = threadIdx.x;
  float s = 0.f;
  for (int k = tid; k < NUM_EMB; k += 256) {
    float p = counts[k] * (1.0f / (float)NROWS);
    s += p * logf(p + 1e-10f);
  }
  red[tid] = s;
  __syncthreads();
  for (int st = 128; st > 0; st >>= 1) {
    if (tid < st) red[tid] += red[tid + st];
    __syncthreads();
  }
  if (tid == 0) {
    out[0]       = 1.25f * sse[0] * (1.0f / (float)BDT);  // q_latent + 0.25*e_latent
    out[1 + BDT] = expf(-red[0]);                          // perplexity
  }
}

// ---- Host launcher ----------------------------------------------------------
extern "C" void kernel_launch(void* const* d_in, const int* in_sizes, int n_in,
                              void* d_out, int out_size, void* d_ws, size_t ws_size,
                              hipStream_t stream) {
  const float* x = (const float*)d_in[0];   // [64, 512, 256] fp32
  const float* W = (const float*)d_in[1];   // [2048, 512] fp32
  float* out = (float*)d_out;

  // Workspace layout (~4.5 MB total)
  char* ws = (char*)d_ws;
  unsigned short* whi  = (unsigned short*)ws;                          // 2 MB
  unsigned short* wlo  = (unsigned short*)(ws + 2097152);              // 2 MB
  float*          cnrm = (float*)(ws + 4194304);                       // 8 KB
  unsigned*       idx3 = (unsigned*)(ws + 4194304 + 8192);             // 192 KB
  float*          cnts = (float*)(ws + 4194304 + 8192 + 196608);       // 8 KB
  float*          sse  = cnts + NUM_EMB;                               // 4 B

  (void)hipFuncSetAttribute((const void*)vq_gemm_top3,
                            hipFuncAttributeMaxDynamicSharedMemorySize, SMEM_BYTES);

  vq_prep_codes<<<NUM_EMB / 8, 256, 0, stream>>>(W, whi, wlo, cnrm);
  vq_init<<<(NUM_EMB + 255) / 256, 256, 0, stream>>>(cnts, sse);
  vq_gemm_top3<<<NROWS / GB_ROWS, GB_THREADS, SMEM_BYTES, stream>>>(x, whi, wlo, cnrm, idx3);
  vq_gather_outputs<<<256, 256, 0, stream>>>(x, W, idx3, out, cnts, sse);
  vq_finalize<<<1, 256, 0, stream>>>(cnts, sse, out);
}
